// BodyPartAttention_45664092291331
// MI455X (gfx1250) — matrix operand order repaired and checked by reading, not compiled, not run on hardware
//
#include <hip/hip_runtime.h>

typedef __attribute__((ext_vector_type(16))) __bf16 v16bf;
typedef __attribute__((ext_vector_type(8)))  __bf16 v8bf;
typedef __attribute__((ext_vector_type(4)))  __bf16 v4bf;
typedef __attribute__((ext_vector_type(8)))  float  v8f;

#define FPB    8               // frames per block
#define PPARTS 6
#define MROWS  (FPB * PPARTS)  // 48 tokens = 3 WMMA M-tiles
#define MT     (MROWS / 16)    // 3
#define DIM    128
#define TD     384
#define NHEADS 4
#define DH     32

__device__ __constant__ int c_idx[6][5] = {
    {0, 1, 2, 3, 4}, {5, 6, 11, 12, -1}, {5, 7, 9, -1, -1},
    {6, 8, 10, -1, -1}, {11, 13, 15, -1, -1}, {12, 14, 16, -1, -1}};
__device__ __constant__ int c_len[6] = {5, 4, 3, 3, 3, 3};

// A fragment: 16x32 bf16 tile, row-major LDS (ld = DIM halves).
// lanes 0-15: M=lane, K 0..15; lanes 16-31: M=lane-16, K 16..31 (contiguous k)
__device__ inline v16bf ld_a_lds(const __bf16* base, int row, int kt, int lane) {
    int kh = (lane & 16) ? 16 : 0;
    const __bf16* p = base + row * DIM + kt * 32 + kh;
    v8bf lo = *(const v8bf*)(p);
    v8bf hi = *(const v8bf*)(p + 8);
    v16bf a;
#pragma unroll
    for (int i = 0; i < 8; ++i) { a[i] = lo[i]; a[i + 8] = hi[i]; }
    return a;
}

// B fragment: 32x16 tile with B[k][n] = W[n][k]; W row-major f32 (ld = DIM).
// lane holds column n; lanes 0-15 hold K=0..15, lanes 16-31 hold K=16..31 (contiguous k).
__device__ inline v16bf ld_b_w(const float* __restrict__ W, int n, int kt, int lane) {
    const float* p = W + n * DIM + kt * 32 + ((lane & 16) ? 16 : 0);
    v16bf b;
#pragma unroll
    for (int i = 0; i < 16; i += 4) {
        float4 x = *(const float4*)(p + i);
        b[i] = (__bf16)x.x; b[i + 1] = (__bf16)x.y;
        b[i + 2] = (__bf16)x.z; b[i + 3] = (__bf16)x.w;
    }
    return b;
}

__global__ __launch_bounds__(256)
void bodypart_attn_kernel(const float* __restrict__ h_joint,
                          const float* __restrict__ mask_joint,
                          const float* __restrict__ W_proj, const float* __restrict__ b_proj,
                          const float* __restrict__ W_in,   const float* __restrict__ b_in,
                          const float* __restrict__ W_out,  const float* __restrict__ b_out,
                          const float* __restrict__ ln_g,   const float* __restrict__ ln_b,
                          float* __restrict__ out_part,     float* __restrict__ out_imp) {
    __shared__ __bf16 s_tok[MROWS * DIM];   // pooled tokens (A of GEMM1)
    __shared__ __bf16 s_qbf[MROWS * DIM];   // q bf16 (A of GEMM2)
    __shared__ float  s_qf [MROWS * DIM];   // q f32 -> later y (residual/LN)
    __shared__ float  s_qkv[MROWS * TD];    // qkv f32
    __shared__ __bf16 s_obf[MROWS * DIM];   // attention out bf16 (A of GEMM3)
    __shared__ float  s_mask[FPB * 17];
    __shared__ float  s_den[MROWS];
    __shared__ float  s_imp[MROWS];

    const int tid  = threadIdx.x;
    const int wave = tid >> 5;
    const int lane = tid & 31;
    const long frame0 = (long)blockIdx.x * FPB;

    // ---- Phase 0: mask flags, denominators, importance init ----
    for (int i = tid; i < FPB * 17; i += 256) {
        int f = i / 17, j = i % 17;
        s_mask[i] = (mask_joint[(frame0 + f) * 17 + j] > 0.05f) ? 1.0f : 0.0f;
    }
    if (tid < MROWS) s_imp[tid] = 0.0f;
    __syncthreads();

    if (tid < MROWS) {
        int f = tid / PPARTS, p = tid % PPARTS;
        float s = 0.0f;
        for (int l = 0; l < c_len[p]; ++l) s += s_mask[f * 17 + c_idx[p][l]];
        s_den[tid] = fmaxf(s, 1e-6f);
    }
    __syncthreads();

    // ---- Phase 1: mask-aware mean pool -> tok (bf16), float4 vectorized ----
    for (int i = tid; i < MROWS * DIM / 4; i += 256) {
        int m = i >> 5, dq = (i & 31) << 2;           // token row, dim/4 chunk
        int f = m / PPARTS, p = m % PPARTS;
        float ax = 0.f, ay = 0.f, az = 0.f, aw = 0.f;
        const int len = c_len[p];
        for (int l = 0; l < len; ++l) {
            int j = c_idx[p][l];
            float mf = s_mask[f * 17 + j];
            float4 hv = *(const float4*)&h_joint[((frame0 + f) * 17 + j) * DIM + dq];
            ax += mf * hv.x; ay += mf * hv.y; az += mf * hv.z; aw += mf * hv.w;
        }
        float inv = 1.0f / s_den[m];
        v4bf o;
        o[0] = (__bf16)(ax * inv); o[1] = (__bf16)(ay * inv);
        o[2] = (__bf16)(az * inv); o[3] = (__bf16)(aw * inv);
        *(v4bf*)&s_tok[m * DIM + dq] = o;
    }
    __syncthreads();

    // ---- Phase 2: GEMM1  q = tok @ Wproj^T + b_proj  (ni = wave; B hoisted) ----
    {
        int n = wave * 16 + (lane & 15);
        v16bf bf[4];
#pragma unroll
        for (int kt = 0; kt < 4; ++kt) bf[kt] = ld_b_w(W_proj, n, kt, lane);
        float bp = b_proj[n];
#pragma unroll
        for (int mi = 0; mi < MT; ++mi) {
            v8f acc = {};
#pragma unroll
            for (int kt = 0; kt < 4; ++kt) {
                v16bf a = ld_a_lds(s_tok, mi * 16 + (lane & 15), kt, lane);
                acc = __builtin_amdgcn_wmma_f32_16x16x32_bf16(false, a, false, bf[kt],
                                                              (short)0, acc, false, false);
            }
            int mBase = mi * 16 + ((lane & 16) ? 8 : 0);
#pragma unroll
            for (int r = 0; r < 8; ++r) {
                float v = acc[r] + bp;
                int m = mBase + r;
                s_qf [m * DIM + n] = v;
                s_qbf[m * DIM + n] = (__bf16)v;
            }
        }
    }
    __syncthreads();

    // ---- Phase 3: GEMM2  qkv = q @ Win^T + b_in  (3 ni per wave; B hoisted) ----
    for (int c = 0; c < 3; ++c) {
        int ni = wave + 8 * c;                       // 0..23
        int n  = ni * 16 + (lane & 15);
        v16bf bf[4];
#pragma unroll
        for (int kt = 0; kt < 4; ++kt) bf[kt] = ld_b_w(W_in, n, kt, lane);
        float bi = b_in[n];
#pragma unroll
        for (int mi = 0; mi < MT; ++mi) {
            v8f acc = {};
#pragma unroll
            for (int kt = 0; kt < 4; ++kt) {
                v16bf a = ld_a_lds(s_qbf, mi * 16 + (lane & 15), kt, lane);
                acc = __builtin_amdgcn_wmma_f32_16x16x32_bf16(false, a, false, bf[kt],
                                                              (short)0, acc, false, false);
            }
            int mBase = mi * 16 + ((lane & 16) ? 8 : 0);
#pragma unroll
            for (int r = 0; r < 8; ++r)
                s_qkv[(mBase + r) * TD + n] = acc[r] + bi;
        }
    }
    __syncthreads();

    // ---- Phase 4: per-(frame,head,qrow) attention over P=6 parts (VALU f32) ----
    if (tid < FPB * NHEADS * PPARTS) {   // 192 threads
        int f   = tid / (NHEADS * PPARTS);
        int rem = tid % (NHEADS * PPARTS);
        int h   = rem / PPARTS;
        int qr  = rem % PPARTS;
        const float4* qp = (const float4*)&s_qkv[(f * PPARTS + qr) * TD + h * DH];
        float4 qv[8];
#pragma unroll
        for (int cc = 0; cc < 8; ++cc) qv[cc] = qp[cc];

        float s[PPARTS], smax = -1e30f;
#pragma unroll
        for (int j = 0; j < PPARTS; ++j) {
            const float4* kp = (const float4*)&s_qkv[(f * PPARTS + j) * TD + DIM + h * DH];
            float d = 0.0f;
#pragma unroll
            for (int cc = 0; cc < 8; ++cc) {
                float4 kv = kp[cc];
                d += qv[cc].x * kv.x + qv[cc].y * kv.y + qv[cc].z * kv.z + qv[cc].w * kv.w;
            }
            s[j] = d * 0.17677669529663687f;   // 1/sqrt(32)
            smax = fmaxf(smax, s[j]);
        }
        float se = 0.0f;
#pragma unroll
        for (int j = 0; j < PPARTS; ++j) { s[j] = expf(s[j] - smax); se += s[j]; }
        float inv = 1.0f / se;
#pragma unroll
        for (int j = 0; j < PPARTS; ++j) {
            s[j] *= inv;
            atomicAdd(&s_imp[f * PPARTS + j], s[j]);
        }
        float4 ov[8];
#pragma unroll
        for (int cc = 0; cc < 8; ++cc) { ov[cc].x = 0.f; ov[cc].y = 0.f; ov[cc].z = 0.f; ov[cc].w = 0.f; }
#pragma unroll
        for (int j = 0; j < PPARTS; ++j) {
            const float4* vp = (const float4*)&s_qkv[(f * PPARTS + j) * TD + 2 * DIM + h * DH];
            float wj = s[j];
#pragma unroll
            for (int cc = 0; cc < 8; ++cc) {
                float4 vv = vp[cc];
                ov[cc].x += wj * vv.x; ov[cc].y += wj * vv.y;
                ov[cc].z += wj * vv.z; ov[cc].w += wj * vv.w;
            }
        }
        __bf16* orow = &s_obf[(f * PPARTS + qr) * DIM + h * DH];
#pragma unroll
        for (int cc = 0; cc < 8; ++cc) {
            v4bf ob;
            ob[0] = (__bf16)ov[cc].x; ob[1] = (__bf16)ov[cc].y;
            ob[2] = (__bf16)ov[cc].z; ob[3] = (__bf16)ov[cc].w;
            *(v4bf*)(orow + cc * 4) = ob;
        }
    }
    __syncthreads();

    // ---- Phase 5: GEMM3  y = q + (o @ Wout^T + b_out)  (ni = wave; B hoisted) ----
    {
        int n = wave * 16 + (lane & 15);
        v16bf bf[4];
#pragma unroll
        for (int kt = 0; kt < 4; ++kt) bf[kt] = ld_b_w(W_out, n, kt, lane);
        float bo = b_out[n];
#pragma unroll
        for (int mi = 0; mi < MT; ++mi) {
            v8f acc = {};
#pragma unroll
            for (int kt = 0; kt < 4; ++kt) {
                v16bf a = ld_a_lds(s_obf, mi * 16 + (lane & 15), kt, lane);
                acc = __builtin_amdgcn_wmma_f32_16x16x32_bf16(false, a, false, bf[kt],
                                                              (short)0, acc, false, false);
            }
            int mBase = mi * 16 + ((lane & 16) ? 8 : 0);
#pragma unroll
            for (int r = 0; r < 8; ++r) {
                int m = mBase + r;
                s_qf[m * DIM + n] = acc[r] + bo + s_qf[m * DIM + n];
            }
        }
    }
    __syncthreads();

    // ---- Phase 6: LayerNorm + writeout (one wave per 6 tokens, float4/lane) ----
    {
        float4 g  = ((const float4*)ln_g)[lane];
        float4 bb = ((const float4*)ln_b)[lane];
        for (int tl = 0; tl < MROWS / 8; ++tl) {
            int m = wave * (MROWS / 8) + tl;
            float4 yv = *(const float4*)&s_qf[m * DIM + lane * 4];
            float s = yv.x + yv.y + yv.z + yv.w;
#pragma unroll
            for (int off = 16; off > 0; off >>= 1) s += __shfl_xor(s, off, 32);
            float mu = s * (1.0f / DIM);
            float tx = yv.x - mu, ty = yv.y - mu, tz = yv.z - mu, tw = yv.w - mu;
            float v = tx * tx + ty * ty + tz * tz + tw * tw;
#pragma unroll
            for (int off = 16; off > 0; off >>= 1) v += __shfl_xor(v, off, 32);
            float rstd = rsqrtf(v * (1.0f / DIM) + 1e-5f);
            float4 o;
            o.x = tx * rstd * g.x + bb.x;
            o.y = ty * rstd * g.y + bb.y;
            o.z = tz * rstd * g.z + bb.z;
            o.w = tw * rstd * g.w + bb.w;
            long obase = (frame0 * PPARTS + m) * DIM;
            *(float4*)&out_part[obase + lane * 4] = o;
        }
    }
    if (tid < MROWS)
        out_imp[frame0 * PPARTS + tid] = s_imp[tid] * (1.0f / (NHEADS * PPARTS));
}

extern "C" void kernel_launch(void* const* d_in, const int* in_sizes, int n_in,
                              void* d_out, int out_size, void* d_ws, size_t ws_size,
                              hipStream_t stream) {
    (void)in_sizes; (void)n_in; (void)out_size; (void)d_ws; (void)ws_size;
    const float* h_joint    = (const float*)d_in[0];
    const float* mask_joint = (const float*)d_in[1];
    const float* W_proj     = (const float*)d_in[2];
    const float* b_proj     = (const float*)d_in[3];
    const float* W_in       = (const float*)d_in[4];
    const float* b_in       = (const float*)d_in[5];
    const float* W_out      = (const float*)d_in[6];
    const float* b_out      = (const float*)d_in[7];
    const float* ln_g       = (const float*)d_in[8];
    const float* ln_b       = (const float*)d_in[9];

    float* out_part = (float*)d_out;
    const long PART_OUT_ELEMS = 32L * 512L * PPARTS * DIM;  // 12,582,912
    float* out_imp = out_part + PART_OUT_ELEMS;

    const int frames = 32 * 512;
    dim3 grid(frames / FPB), block(256);
    hipLaunchKernelGGL(bodypart_attn_kernel, grid, block, 0, stream,
                       h_joint, mask_joint, W_proj, b_proj, W_in, b_in,
                       W_out, b_out, ln_g, ln_b, out_part, out_imp);
}
